// RVLinearlayer_72499047956721
// MI455X (gfx1250) — matrix operand order repaired and checked by reading, not compiled, not run on hardware
//
#include <hip/hip_runtime.h>

// Problem constants: B=64, n=SIZE_IN=1024, m=SIZE_OUT=1024
#define NN 1024
#define MM 1024
#define BB 64

typedef __attribute__((ext_vector_type(2)))  float     v2f;
typedef __attribute__((ext_vector_type(8)))  float     v8f;
typedef __attribute__((ext_vector_type(4)))  _Float16  v4h;
typedef __attribute__((ext_vector_type(8)))  _Float16  v8h;
typedef __attribute__((ext_vector_type(16))) _Float16  v16h;

__device__ __forceinline__ float softplus_eps(float x) {
    float ax = fabsf(x);
    return fmaxf(x, 0.0f) + log1pf(expf(-ax)) + 1e-6f;
}

__device__ __forceinline__ v16h cat8(v8h lo, v8h hi) {
    return __builtin_shufflevector(lo, hi, 0,1,2,3,4,5,6,7,8,9,10,11,12,13,14,15);
}

// ---------------------------------------------------------------------------
// Kernel 0a: convert sigma_in (f32) -> f16 copy in ws (streaming b128 -> b64).
// ---------------------------------------------------------------------------
__global__ __launch_bounds__(256)
void convert_s_kernel(const float* __restrict__ s, _Float16* __restrict__ sh) {
    int t = blockIdx.x * 256 + threadIdx.x;          // x4 elements
    float4 v = ((const float4*)s)[t];
    v4h h = { (_Float16)v.x, (_Float16)v.y, (_Float16)v.z, (_Float16)v.w };
    ((v4h*)sh)[t] = h;
}

// ---------------------------------------------------------------------------
// Kernel 0b: WT[o][i] = (f16) w_mu[i][o]   (transposed f16 weights).
// ---------------------------------------------------------------------------
__global__ __launch_bounds__(256)
void convert_wmuT_kernel(const float* __restrict__ w_mu, _Float16* __restrict__ wT) {
    int t = blockIdx.x * 256 + threadIdx.x;          // 1048576
    int o = t >> 10, i = t & 1023;
    wT[t] = (_Float16)w_mu[i * MM + o];
}

// ---------------------------------------------------------------------------
// Kernel 1: W_Sig = softplus(w_sigma)+1e-6 (to ws) + w_kl partial sums.
// ---------------------------------------------------------------------------
__global__ __launch_bounds__(256)
void prep_wsig_kernel(const float* __restrict__ w_sigma,
                      const float* __restrict__ w_mu,
                      float* __restrict__ wsig,
                      float* __restrict__ partials) {
    __shared__ float red[256];
    int e = blockIdx.x * 256 + threadIdx.x;
    float sp = softplus_eps(w_sigma[e]);
    wsig[e] = sp;
    float wm = w_mu[e];
    red[threadIdx.x] = sp - logf(sp) + wm * wm;
    __syncthreads();
    for (int s = 128; s > 0; s >>= 1) {
        if (threadIdx.x < s) red[threadIdx.x] += red[threadIdx.x + s];
        __syncthreads();
    }
    if (threadIdx.x == 0) partials[blockIdx.x] = red[0];
}

// ---------------------------------------------------------------------------
// Kernel 2: deterministic final reduction -> w_kl, b_kl.
// ---------------------------------------------------------------------------
__global__ __launch_bounds__(256)
void kl_finalize_kernel(const float* __restrict__ partials,
                        const float* __restrict__ b_sigma,
                        const float* __restrict__ b_mu,
                        float* __restrict__ out_wkl,
                        float* __restrict__ out_bkl) {
    __shared__ float red[256];
    int tid = threadIdx.x;
    float s = 0.0f;
    for (int i = tid; i < 4096; i += 256) s += partials[i];
    red[tid] = s;
    __syncthreads();
    for (int st = 128; st > 0; st >>= 1) {
        if (tid < st) red[tid] += red[tid + st];
        __syncthreads();
    }
    if (tid == 0) out_wkl[0] = 0.5f * (red[0] - (float)(NN * MM));
    __syncthreads();
    float sb = 0.0f;
    for (int i = tid; i < MM; i += 256) {
        float bs = softplus_eps(b_sigma[i]);
        float bm = b_mu[i];
        sb += bs - logf(bs) + bm * bm;
    }
    red[tid] = sb;
    __syncthreads();
    for (int st = 128; st > 0; st >>= 1) {
        if (tid < st) red[tid] += red[tid + st];
        __syncthreads();
    }
    if (tid == 0) out_bkl[0] = 0.5f * (red[0] - (float)MM);
}

// ---------------------------------------------------------------------------
// Kernel 3: per (b,o): mu_out, diag_term = trW + quad + B_Sig (f32, exact).
// ---------------------------------------------------------------------------
__global__ __launch_bounds__(256)
void mu_diag_kernel(const float* __restrict__ mu_in,
                    const float* __restrict__ sigma_in,
                    const float* __restrict__ w_mu,
                    const float* __restrict__ wsig,
                    const float* __restrict__ b_mu,
                    const float* __restrict__ b_sigma,
                    float* __restrict__ out_mu,
                    float* __restrict__ diag_term) {
    int t = blockIdx.x * 256 + threadIdx.x;
    int b  = t >> 10;
    int o  = t & (MM - 1);
    int o2 = t >> 6;
    int b2 = t & 63;
    const float* mi    = mu_in + b * NN;
    const float* wrow  = wsig + (size_t)o * NN;
    const float* wrow2 = wsig + (size_t)o2 * NN;
    float mu = 0.0f, quad = 0.0f, trw = 0.0f;
    for (int i = 0; i < NN; ++i) {
        float m = mi[i];
        mu   = fmaf(w_mu[i * MM + o], m, mu);
        quad = fmaf(wrow[i], m * m, quad);
        trw  = fmaf(wrow2[i], sigma_in[i * 65600 + b2], trw);
    }
    out_mu[t]    = mu + b_mu[o];
    diag_term[t] = trw + quad + softplus_eps(b_sigma[o]);
}

// ---------------------------------------------------------------------------
// Kernel 4 (dominant, f16 WMMA): Sigma[b] = w_mu^T S[b] w_mu + diag(dterm).
// T = S * w_mu (stage 1), Out = w_mu^T * T (stage 2); T kept transposed in LDS.
// Unroll pragmas keep the live set under the 256-VGPR cap (128 accum VGPRs are
// persistent) so nothing spills to scratch.
// ---------------------------------------------------------------------------
__global__ __launch_bounds__(512)
void sigma_wmma_f16_kernel(const _Float16* __restrict__ Sh,
                           const _Float16* __restrict__ WT,
                           const float* __restrict__ diag_term,
                           float* __restrict__ out_sigma) {
    __shared__ __align__(16) _Float16 Tlds[64 * 136];   // [p][i] padded, 17408 B

    const int b  = blockIdx.x >> 4;          // 0..63
    const int PB = (blockIdx.x & 15) * 64;   // output column block

    const int lane = threadIdx.x & 31;
    const int wave = threadIdx.x >> 5;       // 0..15
    const int hl   = lane >> 4;              // half-wave select
    const int L    = lane & 15;

    const int s1_it  = wave >> 1;            // i-subtile 0..7
    const int s1_pt0 = (wave & 1) * 2;       // p-subtiles {pt0, pt0+1} share A

    const int obase = wave * 64;             // stage-2: 64 o-rows x 64 p-cols

    const _Float16* Sb = Sh + (size_t)b * (NN * NN);

    v8f acc[16];
    #pragma unroll
    for (int c = 0; c < 16; ++c) acc[c] = {};

    #pragma unroll 1
    for (int ip = 0; ip < 8; ++ip) {
        const int IP = ip * 128;
        __syncthreads();                      // guard LDS reuse

        // ---- stage 1: T[i,p] = sum_j S[b,i,j] * w_mu[j,p]
        {
            v8f t0 = {}, t1 = {};
            const _Float16* arow  = Sb + (size_t)(IP + s1_it * 16 + L) * NN;
            const _Float16* brow0 = WT + (size_t)(PB + s1_pt0 * 16 + L) * NN;
            const _Float16* brow1 = brow0 + 16 * NN;
            #pragma unroll 2
            for (int k = 0; k < 32; ++k) {
                const int j0 = k * 32;
                v16h A = cat8(*(const v8h*)(arow + j0 + 16 * hl),
                              *(const v8h*)(arow + j0 + 16 * hl + 8));
                v16h B0 = cat8(*(const v8h*)(brow0 + j0 + 16 * hl),
                               *(const v8h*)(brow0 + j0 + 16 * hl + 8));
                t0 = __builtin_amdgcn_wmma_f32_16x16x32_f16(
                    false, A, false, B0, (short)0, t0, false, false);
                v16h B1 = cat8(*(const v8h*)(brow1 + j0 + 16 * hl),
                               *(const v8h*)(brow1 + j0 + 16 * hl + 8));
                t1 = __builtin_amdgcn_wmma_f32_16x16x32_f16(
                    false, A, false, B1, (short)0, t1, false, false);
            }
            // pack f32->f16 and store TRANSPOSED: lane holds 8 consecutive i
            v8h p0, p1;
            #pragma unroll
            for (int v = 0; v < 8; ++v) {     // D: lane<16 -> M=v, lane>=16 -> M=v+8
                p0[v] = (_Float16)t0[v];
                p1[v] = (_Float16)t1[v];
            }
            const int ioff = s1_it * 16 + 8 * hl;
            *(v8h*)(Tlds + (s1_pt0 * 16 + L) * 136 + ioff)       = p0;
            *(v8h*)(Tlds + ((s1_pt0 + 1) * 16 + L) * 136 + ioff) = p1;
        }
        __syncthreads();

        // ---- stage 2: acc += WT[o, panel] * T[panel, p]
        #pragma unroll 1
        for (int ks = 0; ks < 4; ++ks) {
            const int i0 = ks * 32;           // panel-relative K base
            v16h Bop[4];
            #pragma unroll
            for (int cs = 0; cs < 4; ++cs) {
                const _Float16* tp = Tlds + (cs * 16 + L) * 136 + i0 + 16 * hl;
                Bop[cs] = cat8(*(const v8h*)tp, *(const v8h*)(tp + 8));
            }
            #pragma unroll
            for (int rs = 0; rs < 4; ++rs) {
                const _Float16* ap = WT + (size_t)(obase + rs * 16 + L) * NN + IP + i0;
                v16h A = cat8(*(const v8h*)(ap + 16 * hl),
                              *(const v8h*)(ap + 16 * hl + 8));
                #pragma unroll
                for (int cs = 0; cs < 4; ++cs)
                    acc[rs * 4 + cs] = __builtin_amdgcn_wmma_f32_16x16x32_f16(
                        false, A, false, Bop[cs], (short)0, acc[rs * 4 + cs], false, false);
            }
        }
    }

    // ---- epilogue: diagonal add + store
    const float* dt = diag_term + b * MM;
    float* outB = out_sigma + (size_t)b * (MM * MM);
    #pragma unroll
    for (int rs = 0; rs < 4; ++rs) {
        float dvals[8];
        #pragma unroll
        for (int v = 0; v < 8; ++v) dvals[v] = dt[obase + rs * 16 + v + 8 * hl];
        #pragma unroll
        for (int cs = 0; cs < 4; ++cs)
            #pragma unroll
            for (int v = 0; v < 8; ++v) {
                int o_g = obase + rs * 16 + v + 8 * hl;
                int p_g = PB + cs * 16 + L;
                float val = acc[rs * 4 + cs][v];
                if (o_g == p_g) val += dvals[v];
                outB[(size_t)o_g * MM + p_g] = val;
            }
    }
}

// ---------------------------------------------------------------------------
// Fallback (small ws): fused f32 WMMA version (needs only ~4.5 MB ws).
// ---------------------------------------------------------------------------
__global__ __launch_bounds__(512)
void sigma_wmma_f32_kernel(const float* __restrict__ sigma_in,
                           const float* __restrict__ w_mu,
                           const float* __restrict__ diag_term,
                           float* __restrict__ out_sigma) {
    __shared__ float Vp[64 * 132];
    const int b       = blockIdx.x >> 4;
    const int RB      = (blockIdx.x & 15) * 64;
    const int lane = threadIdx.x & 31;
    const int wave = threadIdx.x >> 5;
    const int hl   = lane >> 4;
    const int L    = lane & 15;
    const int s1_ro  = wave >> 2;
    const int s1_co0 = (wave & 3) * 2;
    const int rt = wave >> 2;
    const int ct = wave & 3;
    const float* Sb = sigma_in + (size_t)b * (NN * NN);
    v8f accO[16];
    #pragma unroll
    for (int c = 0; c < 16; ++c) accO[c] = {};
    #pragma unroll 1
    for (int jp = 0; jp < 8; ++jp) {
        const int JB = jp * 128;
        __syncthreads();
        {
            v8f acc0 = {}, acc1 = {};
            const int acol  = RB + s1_ro * 16 + L;
            const int bcol0 = JB + s1_co0 * 16 + L;
            const int bcol1 = bcol0 + 16;
            #pragma unroll 2
            for (int ki = 0; ki < 256; ++ki) {
                const int i0 = ki * 4 + 2 * hl;
                v2f a, b0, b1;
                a.x  = w_mu[i0 * MM + acol];
                a.y  = w_mu[(i0 + 1) * MM + acol];
                b0.x = Sb[(size_t)i0 * NN + bcol0];
                b0.y = Sb[(size_t)(i0 + 1) * NN + bcol0];
                b1.x = Sb[(size_t)i0 * NN + bcol1];
                b1.y = Sb[(size_t)(i0 + 1) * NN + bcol1];
                acc0 = __builtin_amdgcn_wmma_f32_16x16x4_f32(false, a, false, b0, (short)0, acc0, false, false);
                acc1 = __builtin_amdgcn_wmma_f32_16x16x4_f32(false, a, false, b1, (short)0, acc1, false, false);
            }
            #pragma unroll
            for (int v = 0; v < 8; ++v) {
                int r = s1_ro * 16 + v + hl * 8;
                Vp[r * 132 + s1_co0 * 16 + L]       = acc0[v];
                Vp[r * 132 + (s1_co0 + 1) * 16 + L] = acc1[v];
            }
        }
        __syncthreads();
        #pragma unroll 1
        for (int kj = 0; kj < 32; ++kj) {
            const int j0 = kj * 4 + 2 * hl;
            v2f a;
            a.x = Vp[(rt * 16 + L) * 132 + j0];
            a.y = Vp[(rt * 16 + L) * 132 + j0 + 1];
            const int grow0 = (JB + j0) * MM;
            const int grow1 = grow0 + MM;
            #pragma unroll
            for (int cc = 0; cc < 16; ++cc) {
                const int pcol = ct * 256 + cc * 16 + L;
                v2f bb;
                bb.x = w_mu[grow0 + pcol];
                bb.y = w_mu[grow1 + pcol];
                accO[cc] = __builtin_amdgcn_wmma_f32_16x16x4_f32(false, a, false, bb, (short)0, accO[cc], false, false);
            }
        }
    }
    const float* dt = diag_term + b * MM;
    float* outB = out_sigma + (size_t)b * (MM * MM);
    #pragma unroll
    for (int cc = 0; cc < 16; ++cc)
        #pragma unroll
        for (int v = 0; v < 8; ++v) {
            int o_g = RB + rt * 16 + v + hl * 8;
            int p_g = ct * 256 + cc * 16 + L;
            float val = accO[cc][v];
            if (o_g == p_g) val += dt[o_g];
            outB[(size_t)o_g * MM + p_g] = val;
        }
}

// ---------------------------------------------------------------------------
// Launch. d_in: mu_in, sigma_in, w_mu, w_sigma, b_mu, b_sigma
// d_out: [mu_out 65536][Sigma_out 67108864][w_kl][b_kl]
// ws (f16 path, ~141 MB): [S_h 128MB][WT 2MB][wsig 4MB][dterm 256KB][part 16KB]
// ---------------------------------------------------------------------------
extern "C" void kernel_launch(void* const* d_in, const int* in_sizes, int n_in,
                              void* d_out, int out_size, void* d_ws, size_t ws_size,
                              hipStream_t stream) {
    (void)in_sizes; (void)n_in; (void)out_size;
    const float* mu_in    = (const float*)d_in[0];
    const float* sigma_in = (const float*)d_in[1];
    const float* w_mu     = (const float*)d_in[2];
    const float* w_sigma  = (const float*)d_in[3];
    const float* b_mu     = (const float*)d_in[4];
    const float* b_sigma  = (const float*)d_in[5];

    float* out = (float*)d_out;
    float* out_mu  = out;
    float* out_sig = out + 65536;
    float* out_wkl = out + 65536 + 67108864;
    float* out_bkl = out_wkl + 1;

    const size_t SH_BYTES = (size_t)BB * NN * NN * sizeof(_Float16);   // 128 MB
    const size_t WT_BYTES = (size_t)MM * NN * sizeof(_Float16);        // 2 MB
    const size_t F16_NEED = SH_BYTES + WT_BYTES + (1048576 + 65536 + 4096) * sizeof(float);

    char* wsb = (char*)d_ws;
    bool use_f16 = (ws_size >= F16_NEED);

    float* wsig;
    float* dterm;
    float* partial;
    _Float16* Sh = nullptr;
    _Float16* WT = nullptr;
    if (use_f16) {
        Sh   = (_Float16*)wsb;
        WT   = (_Float16*)(wsb + SH_BYTES);
        wsig = (float*)(wsb + SH_BYTES + WT_BYTES);
    } else {
        wsig = (float*)wsb;
    }
    dterm   = wsig + 1048576;
    partial = dterm + 65536;

    if (use_f16) {
        convert_s_kernel<<<65536, 256, 0, stream>>>(sigma_in, Sh);
        convert_wmuT_kernel<<<4096, 256, 0, stream>>>(w_mu, WT);
    }
    prep_wsig_kernel<<<4096, 256, 0, stream>>>(w_sigma, w_mu, wsig, partial);
    kl_finalize_kernel<<<1, 256, 0, stream>>>(partial, b_sigma, b_mu, out_wkl, out_bkl);
    mu_diag_kernel<<<256, 256, 0, stream>>>(mu_in, sigma_in, w_mu, wsig, b_mu, b_sigma,
                                            out_mu, dterm);
    if (use_f16) {
        sigma_wmma_f16_kernel<<<1024, 512, 0, stream>>>(Sh, WT, dterm, out_sig);
    } else {
        sigma_wmma_f32_kernel<<<1024, 512, 0, stream>>>(sigma_in, w_mu, dterm, out_sig);
    }
}
